// Model_22746146799733
// MI455X (gfx1250) — matrix operand order repaired and checked by reading, not compiled, not run on hardware
//
#include <hip/hip_runtime.h>
#include <hip/hip_bf16.h>

// ---------------- problem constants ----------------
#define N_NODES   50000
#define N_EDGES   250000
#define N_GRAPHS  1000
#define D_RAW     10
#define D_HID     300
#define D_OUT     31

#define DPAD      320                 // 300 padded to 32*10
#define NPAD      50016               // 50000 padded to 16
#define MT        (NPAD / 16)         // 3126 row tiles
#define KT        (DPAD / 32)         // 10 K-tiles (of 32, bf16 WMMA)
#define NT        (DPAD / 16)         // 20 N-tiles (of 16)
#define SBUF      ((size_t)NPAD * DPAD * sizeof(float))   // one activation buffer
#define WPACK_HALFS (KT * NT * 32 * 16)                    // 102400 bf16 values
#define WPACK_BYTES (WPACK_HALFS * 2)                      // 204800 B per matrix
#define SLAB_Q    (NT * 32 * 2)                            // uint4 per K-slab = 1280 (20 KB)

typedef __bf16 v16bf __attribute__((ext_vector_type(16)));
typedef float  v8f   __attribute__((ext_vector_type(8)));

__device__ __forceinline__ unsigned short f2bf(float f) {
    unsigned u = __builtin_bit_cast(unsigned, f);
    u += 0x7fffu + ((u >> 16) & 1u);      // round-to-nearest-even
    return (unsigned short)(u >> 16);
}

union Frag16 {
    v16bf          v;
    unsigned short h[16];
    uint4          q[2];
};

// ---------------- weight repack: fp32 [300,300] -> bf16 WMMA B-fragments ----------------
// Packed layout (ushort flat index): ((kt*NT + nt)*32 + lane)*16 + i
//   lane: col n = nt*16 + (lane&15), h = lane>>4
//   element i = K kt*32 + (i<8 ? 8h+i : 16+8h+(i-8)); even-K in low 16 bits of each dword.
// One K-slab (fixed kt) is a contiguous 20 KB block -> staged to LDS by the GEMM.
__global__ void pack_weight_bf16(const float* __restrict__ W,
                                 unsigned short* __restrict__ Wp) {
    int idx = blockIdx.x * blockDim.x + threadIdx.x;
    if (idx >= WPACK_HALFS) return;
    int i    = idx & 15;
    int lane = (idx >> 4) & 31;
    int tile = idx >> 9;                  // kt*NT + nt
    int nt   = tile % NT;
    int kt   = tile / NT;
    int n = nt * 16 + (lane & 15);
    int k = kt * 32 + (i < 8 ? ((lane >> 4) * 8) + i : 16 + ((lane >> 4) * 8) + (i - 8));
    float v = (k < D_HID && n < D_HID) ? W[k * D_HID + n] : 0.0f;
    Wp[idx] = f2bf(v);
}

// ---------------- lift: h = x @ W_lift + b (K=10, scalar fp32) ----------------
__global__ void lift_kernel(const float* __restrict__ x,
                            const float* __restrict__ Wl,
                            const float* __restrict__ bl,
                            float* __restrict__ h) {
    int idx = blockIdx.x * blockDim.x + threadIdx.x;
    if (idx >= NPAD * DPAD) return;
    int row = idx / DPAD;
    int col = idx - row * DPAD;
    float v = 0.0f;
    if (row < N_NODES && col < D_HID) {
        v = bl[col];
        const float* xr = x + row * D_RAW;
        #pragma unroll
        for (int r = 0; r < D_RAW; ++r) v = fmaf(xr[r], Wl[r * D_HID + col], v);
    }
    h[idx] = v;
}

// ---------------- C = relu(A @ W + bias) ----------------
// Block = 256 threads (8 waves). Wave w owns row-tile (blockIdx.x*8 + w) across ALL
// 20 N-tiles (20 x v8f accumulators). Weights staged K-slab by K-slab into a
// double-buffered 40 KB LDS region. Software pipeline (enforced with
// sched_group_barrier so the allocator must keep 3 B fragments in flight):
//   6 DS-reads (3 frags) ; 17 x (1 WMMA + 2 DS-reads) ; 3 WMMA drain
// A tiles for kt+1 and slab kt+1 global loads are issued before the WMMA block.
__global__ void __launch_bounds__(256)
gemm_relu_wmma(const float* __restrict__ A,
               const unsigned short* __restrict__ Wp,
               const float* __restrict__ bias,
               float* __restrict__ C) {
    __shared__ uint4 ldsW[2][SLAB_Q];           // 2 x 20 KB

    const int tid   = threadIdx.x;
    const int wave  = tid >> 5;
    const int lane  = tid & 31;
    const int mtile = blockIdx.x * 8 + wave;
    const bool live = (mtile < MT);             // wave-uniform
    const int half  = lane >> 4;
    const int arow  = live ? (mtile * 16 + (lane & 15)) : 0;
    const float* aprow = A + (size_t)arow * DPAD + half * 8;

    const uint4* gW = (const uint4*)Wp;

    // ---- stage slab kt=0 + preload A tile kt=0 ----
    #pragma unroll
    for (int j = 0; j < 5; ++j)
        ldsW[0][j * 256 + tid] = gW[j * 256 + tid];

    float4 fcur[4], fnxt[4];
    fcur[0] = *(const float4*)(aprow + 0);
    fcur[1] = *(const float4*)(aprow + 4);
    fcur[2] = *(const float4*)(aprow + 16);
    fcur[3] = *(const float4*)(aprow + 20);
    __syncthreads();

    v8f acc[NT];
    #pragma unroll
    for (int t = 0; t < NT; ++t) acc[t] = v8f{};

    uint4 stage[5];
    for (int kt = 0; kt < KT; ++kt) {
        const int cur = kt & 1;

        // --- convert current A tile (deps: fcur only, loaded a full iteration ago)
        Frag16 a;
        a.h[0]  = f2bf(fcur[0].x); a.h[1]  = f2bf(fcur[0].y);
        a.h[2]  = f2bf(fcur[0].z); a.h[3]  = f2bf(fcur[0].w);
        a.h[4]  = f2bf(fcur[1].x); a.h[5]  = f2bf(fcur[1].y);
        a.h[6]  = f2bf(fcur[1].z); a.h[7]  = f2bf(fcur[1].w);
        a.h[8]  = f2bf(fcur[2].x); a.h[9]  = f2bf(fcur[2].y);
        a.h[10] = f2bf(fcur[2].z); a.h[11] = f2bf(fcur[2].w);
        a.h[12] = f2bf(fcur[3].x); a.h[13] = f2bf(fcur[3].y);
        a.h[14] = f2bf(fcur[3].z); a.h[15] = f2bf(fcur[3].w);

        // --- issue next-iteration global loads (slab first, then A) ---
        if (kt + 1 < KT) {
            const uint4* gs = gW + (size_t)(kt + 1) * SLAB_Q;
            #pragma unroll
            for (int j = 0; j < 5; ++j) stage[j] = gs[j * 256 + tid];
            const float* an = aprow + (kt + 1) * 32;
            fnxt[0] = *(const float4*)(an + 0);
            fnxt[1] = *(const float4*)(an + 4);
            fnxt[2] = *(const float4*)(an + 16);
            fnxt[3] = *(const float4*)(an + 20);
        }

        // --- 20 WMMAs fed by a 4-deep DS-load ring ---
        if (live) {
            const uint4* bl0 = &ldsW[cur][lane * 2];
            Frag16 bb[4];
            #pragma unroll
            for (int p = 0; p < 3; ++p) {
                bb[p].q[0] = bl0[p * 64];
                bb[p].q[1] = bl0[p * 64 + 1];
            }
            #pragma unroll
            for (int nt = 0; nt < NT; ++nt) {
                if (nt + 3 < NT) {
                    Frag16& w = bb[(nt + 3) & 3];
                    w.q[0] = bl0[(nt + 3) * 64];
                    w.q[1] = bl0[(nt + 3) * 64 + 1];
                }
                acc[nt] = __builtin_amdgcn_wmma_f32_16x16x32_bf16(
                    false, a.v, false, bb[nt & 3].v, (short)0, acc[nt], false, false);
            }
#if __has_builtin(__builtin_amdgcn_sched_group_barrier)
            // Pin the pipeline: 0x100 = DS read, 0x008 = matrix (WMMA) ops.
            __builtin_amdgcn_sched_group_barrier(0x100, 6, 0);   // 3 frags ahead
            #pragma unroll
            for (int i = 0; i < NT - 3; ++i) {
                __builtin_amdgcn_sched_group_barrier(0x008, 1, 0);
                __builtin_amdgcn_sched_group_barrier(0x100, 2, 0);
            }
            __builtin_amdgcn_sched_group_barrier(0x008, 3, 0);   // drain
#endif
        }

        // --- commit next slab to LDS, rotate A registers ---
        if (kt + 1 < KT) {
            const int nxt = (kt + 1) & 1;
            #pragma unroll
            for (int j = 0; j < 5; ++j) ldsW[nxt][j * 256 + tid] = stage[j];
            __syncthreads();
            #pragma unroll
            for (int j = 0; j < 4; ++j) fcur[j] = fnxt[j];
        }
    }

    if (!live) return;

    // epilogue: D layout lane L -> col-in-tile = (L&15), rows = half*8 + r
    const int colin = lane & 15;
    const int rbase = mtile * 16 + half * 8;
    #pragma unroll
    for (int nt = 0; nt < NT; ++nt) {
        int col = nt * 16 + colin;
        float bb = (col < D_HID) ? bias[col] : 0.0f;
        #pragma unroll
        for (int r = 0; r < 8; ++r) {
            float v = acc[nt][r] + bb;
            C[(size_t)(rbase + r) * DPAD + col] = fmaxf(v, 0.0f);
        }
    }
}

// ---------------- zero helper ----------------
__global__ void zero4_kernel(float4* __restrict__ p, int n4) {
    int i = blockIdx.x * blockDim.x + threadIdx.x;
    if (i < n4) p[i] = make_float4(0.f, 0.f, 0.f, 0.f);
}

// ---------------- edge scatter-add: agg[dst] += msg[src] ----------------
// ReLU output is ~half zeros: skip those atomics to cut L2 atomic traffic.
__global__ void scatter_kernel(const float* __restrict__ msg,
                               const int* __restrict__ src,
                               const int* __restrict__ dst,
                               float* __restrict__ agg) {
    int idx = blockIdx.x * blockDim.x + threadIdx.x;
    if (idx >= N_EDGES * 80) return;          // 80 float4 per 320-wide row
    int e = idx / 80;
    int q = idx - e * 80;
    int s = src[e], d = dst[e];
    float4 v = *(const float4*)(msg + (size_t)s * DPAD + q * 4);
    float* o = agg + (size_t)d * DPAD + q * 4;
    if (v.x != 0.f) atomicAdd(o + 0, v.x);
    if (v.y != 0.f) atomicAdd(o + 1, v.y);
    if (v.z != 0.f) atomicAdd(o + 2, v.z);
    if (v.w != 0.f) atomicAdd(o + 3, v.w);
}

// ---------------- readout + graph segment-sum ----------------
__global__ void readout_kernel(const float* __restrict__ h,
                               const float* __restrict__ Wro,
                               const float* __restrict__ bro,
                               const int* __restrict__ gid,
                               float* __restrict__ out) {
    int idx = blockIdx.x * blockDim.x + threadIdx.x;
    if (idx >= N_NODES * D_OUT) return;
    int n = idx / D_OUT;
    int o = idx - n * D_OUT;
    float v = bro[o];
    const float* hr = h + (size_t)n * DPAD;
    #pragma unroll 4
    for (int k = 0; k < D_HID; ++k) v = fmaf(hr[k], Wro[k * D_OUT + o], v);
    atomicAdd(&out[gid[n] * D_OUT + o], v);
}

// ---------------- launch ----------------
extern "C" void kernel_launch(void* const* d_in, const int* in_sizes, int n_in,
                              void* d_out, int out_size, void* d_ws, size_t ws_size,
                              hipStream_t stream) {
    const float* x   = (const float*)d_in[0];
    const int*   src = (const int*)d_in[1];
    const int*   dst = (const int*)d_in[2];
    const int*   gid = (const int*)d_in[3];
    const float* Wl  = (const float*)d_in[4];
    const float* bl  = (const float*)d_in[5];
    const float* Wro = (const float*)d_in[6];
    const float* bro = (const float*)d_in[7];
    const float* Wc[3] = {(const float*)d_in[8],  (const float*)d_in[12], (const float*)d_in[16]};
    const float* bc[3] = {(const float*)d_in[9],  (const float*)d_in[13], (const float*)d_in[17]};
    const float* Wf[3] = {(const float*)d_in[10], (const float*)d_in[14], (const float*)d_in[18]};
    const float* bf[3] = {(const float*)d_in[11], (const float*)d_in[15], (const float*)d_in[19]};

    char* ws = (char*)d_ws;
    float* buf0 = (float*)ws;
    float* buf1 = (float*)(ws + SBUF);
    unsigned short* wp[6];
    for (int m = 0; m < 6; ++m)
        wp[m] = (unsigned short*)(ws + 2 * SBUF + (size_t)m * WPACK_BYTES);

    const int TB = 256;
    // pack weights: order Wc1,Wf1,Wc2,Wf2,Wc3,Wf3
    int packBlocks = (WPACK_HALFS + TB - 1) / TB;
    for (int l = 0; l < 3; ++l) {
        pack_weight_bf16<<<packBlocks, TB, 0, stream>>>(Wc[l], wp[2 * l]);
        pack_weight_bf16<<<packBlocks, TB, 0, stream>>>(Wf[l], wp[2 * l + 1]);
    }

    lift_kernel<<<(NPAD * DPAD + TB - 1) / TB, TB, 0, stream>>>(x, Wl, bl, buf0);

    dim3 ggrid((MT + 7) / 8);                   // 391 blocks, 8 waves each (1 row-tile/wave)
    int n4buf = (int)(SBUF / sizeof(float4));
    int zBlocks = (n4buf + TB - 1) / TB;
    int sBlocks = (N_EDGES * 80 + TB - 1) / TB;

    float* cur = buf0;
    float* oth = buf1;
    for (int l = 0; l < 3; ++l) {
        gemm_relu_wmma<<<ggrid, 256, 0, stream>>>(cur, wp[2 * l], bc[l], oth);      // msg
        zero4_kernel<<<zBlocks, TB, 0, stream>>>((float4*)cur, n4buf);              // agg = 0
        scatter_kernel<<<sBlocks, TB, 0, stream>>>(oth, src, dst, cur);             // agg += msg[src] at dst
        gemm_relu_wmma<<<ggrid, 256, 0, stream>>>(cur, wp[2 * l + 1], bf[l], oth);  // h = relu(agg@Wf+b)
        float* t = cur; cur = oth; oth = t;
    }

    // output: zero then segment-sum of node logits (31000 floats = 7750 float4)
    zero4_kernel<<<(7750 + TB - 1) / TB, TB, 0, stream>>>((float4*)d_out, 7750);
    readout_kernel<<<(N_NODES * D_OUT + TB - 1) / TB, TB, 0, stream>>>(
        cur, Wro, bro, gid, (float*)d_out);
}
